// MultiHeadMaskedAttention_34454227649231
// MI455X (gfx1250) — matrix operand order repaired and checked by reading, not compiled, not run on hardware
//
#include <hip/hip_runtime.h>
#include <hip/hip_bf16.h>

// ---------- types ----------
typedef __attribute__((ext_vector_type(16))) __bf16        v16bf;
typedef __attribute__((ext_vector_type(8)))  float         v8f;
typedef __attribute__((ext_vector_type(4)))  unsigned int  u32x4;
typedef int v4i_vs __attribute__((vector_size(16)));        // matches builtin param type

union Frag16 { v16bf bf; u32x4 u4[2]; };

__device__ __forceinline__ unsigned short f2bf(float f) {
  union { float f; unsigned u; } v; v.f = f;
  unsigned r = v.u + 0x7FFFu + ((v.u >> 16) & 1u);   // round-to-nearest-even
  return (unsigned short)(r >> 16);
}
__device__ __forceinline__ float bf2f(unsigned short h) {
  union { unsigned u; float f; } v; v.u = ((unsigned)h) << 16;
  return v.f;
}
__device__ __forceinline__ v8f v8zero() {
  v8f z = {0.f,0.f,0.f,0.f,0.f,0.f,0.f,0.f};
  return z;
}

// ---------- CDNA5 async global->LDS path (guarded; sync fallback) ----------
#if __has_builtin(__builtin_amdgcn_global_load_async_to_lds_b128) && \
    __has_builtin(__builtin_amdgcn_s_wait_asynccnt)
#define ASYNC_LDS 1
#else
#define ASYNC_LDS 0
#endif

__device__ __forceinline__ void copy16(unsigned short* l, const unsigned short* g) {
#if ASYNC_LDS
  __builtin_amdgcn_global_load_async_to_lds_b128(
      (__attribute__((address_space(1))) v4i_vs*)(void*)g,
      (__attribute__((address_space(3))) v4i_vs*)(void*)l, 0, 0);
#else
  *(u32x4*)l = *(const u32x4*)g;
#endif
}

// ---------- problem constants ----------
#define GB   2
#define GT   2048
#define GC   4096
#define GH   32
#define GKVH 8
#define GD   128
#define QKVO 6144          // (H + 2*KVH) * D

// =====================================================================
// fp32 -> bf16 conversion (4 elements / thread; n divisible by 1024)
// =====================================================================
__global__ __launch_bounds__(256)
void cvt_f32_bf16(const float* __restrict__ in, unsigned short* __restrict__ out) {
  long long i0 = ((long long)blockIdx.x * 256 + threadIdx.x) * 4;
  float4 v = *(const float4*)(in + i0);
  out[i0 + 0] = f2bf(v.x);
  out[i0 + 1] = f2bf(v.y);
  out[i0 + 2] = f2bf(v.z);
  out[i0 + 3] = f2bf(v.w);
}

// =====================================================================
// Tiled bf16 WMMA GEMM:  C[M,N] = A[M,K] @ B[N,K]^T + bias
// Block tile 128(M) x 128(N), K-step 32; 8 waves as 2(M) x 4(N);
// each wave computes 64x32 = 4x2 tiles of v_wmma_f32_16x16x32_bf16.
// Double-buffered LDS; next K-tile staged with async global->LDS copies
// (s_wait_asynccnt pipelined) while WMMAs consume the current tile.
// =====================================================================
__global__ __launch_bounds__(256)
void gemm_bf16(const unsigned short* __restrict__ A,
               const unsigned short* __restrict__ Bm,
               const float* __restrict__ bias,
               void* __restrict__ Cout,
               int M, int N, int K, int out_bf16)
{
  __shared__ __align__(16) unsigned short lA[2][128 * 32];  // 2 x 8 KB
  __shared__ __align__(16) unsigned short lB[2][128 * 32];  // 2 x 8 KB

  const int tid  = threadIdx.x;
  const int lane = tid & 31;
  const int wave = tid >> 5;
  const int wm   = wave & 1;      // 2 M-blocks of 64 rows
  const int wn   = wave >> 1;     // 4 N-blocks of 32 cols
  const int bm   = blockIdx.y * 128;
  const int bn   = blockIdx.x * 128;
  const int kh   = lane >> 4;     // lane half selects K sub-range

  // staging coords: thread copies row (tid>>1), 16-elem half (tid&1),
  // as two 8-elem (16 B) chunks, for both A and B tiles (4 copies/thread)
  const int sr = tid >> 1;
  const int sc = (tid & 1) * 16;

  v8f acc[4][2];
  for (int a = 0; a < 4; ++a) for (int c = 0; c < 2; ++c) acc[a][c] = v8zero();

  const int nk = K >> 5;

  // prologue: stage K-tile 0 into buffer 0
  {
    copy16(&lA[0][sr * 32 + sc],     &A[(size_t)(bm + sr) * K + sc]);
    copy16(&lA[0][sr * 32 + sc + 8], &A[(size_t)(bm + sr) * K + sc + 8]);
    copy16(&lB[0][sr * 32 + sc],     &Bm[(size_t)(bn + sr) * K + sc]);
    copy16(&lB[0][sr * 32 + sc + 8], &Bm[(size_t)(bn + sr) * K + sc + 8]);
  }

  for (int kt = 0; kt < nk; ++kt) {
    const int cur = kt & 1;
    if (kt + 1 < nk) {
      // stage next K-tile into the other buffer while we compute
      const int k1 = (kt + 1) * 32;
      copy16(&lA[cur ^ 1][sr * 32 + sc],     &A[(size_t)(bm + sr) * K + k1 + sc]);
      copy16(&lA[cur ^ 1][sr * 32 + sc + 8], &A[(size_t)(bm + sr) * K + k1 + sc + 8]);
      copy16(&lB[cur ^ 1][sr * 32 + sc],     &Bm[(size_t)(bn + sr) * K + k1 + sc]);
      copy16(&lB[cur ^ 1][sr * 32 + sc + 8], &Bm[(size_t)(bn + sr) * K + k1 + sc + 8]);
#if ASYNC_LDS
      __builtin_amdgcn_s_wait_asynccnt(4);  // 4 newest in flight -> current tile landed
#endif
      if (kt + 2 < nk) {
        __builtin_prefetch(&A[(size_t)(bm + sr) * K + k1 + 32], 0, 3);
        __builtin_prefetch(&Bm[(size_t)(bn + sr) * K + k1 + 32], 0, 3);
      }
    } else {
#if ASYNC_LDS
      __builtin_amdgcn_s_wait_asynccnt(0);
#endif
    }
    __syncthreads();   // current buffer ready for all waves

    // ---- fragments + 8 WMMAs ----
    Frag16 fa[4], fb[2];
    for (int mt = 0; mt < 4; ++mt) {
      int r = wm * 64 + mt * 16 + (lane & 15);
      fa[mt].u4[0] = *(const u32x4*)&lA[cur][r * 32 + kh * 16];       // K 0..15 / 16..31
      fa[mt].u4[1] = *(const u32x4*)&lA[cur][r * 32 + kh * 16 + 8];
    }
    for (int nt = 0; nt < 2; ++nt) {
      int c = wn * 32 + nt * 16 + (lane & 15);
      fb[nt].u4[0] = *(const u32x4*)&lB[cur][c * 32 + kh * 16];       // K 0..15 / 16..31
      fb[nt].u4[1] = *(const u32x4*)&lB[cur][c * 32 + kh * 16 + 8];
    }
    for (int mt = 0; mt < 4; ++mt)
      for (int nt = 0; nt < 2; ++nt)
        acc[mt][nt] = __builtin_amdgcn_wmma_f32_16x16x32_bf16(
            false, fa[mt].bf, false, fb[nt].bf, (short)0, acc[mt][nt],
            false, false);

    __syncthreads();   // all waves done reading `cur` before it is overwritten
  }

  // ---- epilogue: bias + store (fp32 or bf16) ----
  for (int mt = 0; mt < 4; ++mt)
    for (int nt = 0; nt < 2; ++nt)
      for (int i = 0; i < 8; ++i) {
        int row = bm + wm * 64 + mt * 16 + (lane >> 4) * 8 + i;
        int col = bn + wn * 32 + nt * 16 + (lane & 15);
        float v = acc[mt][nt][i] + (bias ? bias[col] : 0.f);
        if (out_bf16)
          ((unsigned short*)Cout)[(size_t)row * N + col] = f2bf(v);
        else
          ((float*)Cout)[(size_t)row * N + col] = v;
      }
}

// =====================================================================
// RoPE + scatter: qkv(bf16 [B,T,6144]) ->
//   q bf16 [B,H,T,D]   (rotated, pre-scaled by 1/sqrt(D))
//   k bf16 [B,KVH,T,D] (rotated)
//   vT bf16 [B,KVH,D,T] (transposed for contiguous-K WMMA B-fragments)
// Reference RoPE angles depend only on head-dim index: ang = 2i/theta.
// =====================================================================
__global__ __launch_bounds__(256)
void rope_scatter(const unsigned short* __restrict__ qkv,
                  unsigned short* __restrict__ q,
                  unsigned short* __restrict__ k,
                  unsigned short* __restrict__ vt)
{
  int idx = blockIdx.x * 256 + threadIdx.x;     // B*T*3072 total
  int bt  = idx / 3072;
  int u   = idx % 3072;
  int b   = bt >> 11;       // / 2048
  int t   = bt & 2047;
  const unsigned short* row = qkv + (size_t)bt * QKVO;
  const float scale = 0.08838834764831845f;     // 1/sqrt(128)

  if (u < 2048) {                               // Q: 32 heads * 64 pairs
    int h2 = u >> 6, i = u & 63;
    float ang = (float)(2 * i) * 1e-4f;
    float co = cosf(ang), si = sinf(ang);
    float x1 = bf2f(row[h2 * 128 + i]);
    float x2 = bf2f(row[h2 * 128 + 64 + i]);
    size_t base = ((size_t)(b * GH + h2) * GT + t) * GD;
    q[base + i]      = f2bf((x1 * co - x2 * si) * scale);
    q[base + 64 + i] = f2bf((x2 * co + x1 * si) * scale);
  } else if (u < 2560) {                        // K: 8 heads * 64 pairs
    int u2 = u - 2048;
    int h2 = u2 >> 6, i = u2 & 63;
    float ang = (float)(2 * i) * 1e-4f;
    float co = cosf(ang), si = sinf(ang);
    float x1 = bf2f(row[4096 + h2 * 128 + i]);
    float x2 = bf2f(row[4096 + h2 * 128 + 64 + i]);
    size_t base = ((size_t)(b * GKVH + h2) * GT + t) * GD;
    k[base + i]      = f2bf(x1 * co - x2 * si);
    k[base + 64 + i] = f2bf(x2 * co + x1 * si);
  } else {                                      // V transpose: 8 heads * 64 (2 elems each)
    int u2 = u - 2560;
    int h2 = u2 >> 6, i = u2 & 63;
    size_t base = (size_t)(b * GKVH + h2) * GD;
    vt[(base + i) * GT + t]      = row[5120 + h2 * 128 + i];
    vt[(base + 64 + i) * GT + t] = row[5120 + h2 * 128 + 64 + i];
  }
}

// =====================================================================
// Flash attention, causal, GQA (4 Q heads per KV head).
// One workgroup = (b, h, 64-row Q tile); KV tiles of 128.
// 8 waves as 2(M) x 4(N); each wave owns 32x32 of S / O.
// Q and P staged in LDS; K/V fragments streamed from global (L2-resident).
// Online softmax: 4 lanes per row, __shfl_xor butterfly reductions.
// =====================================================================
__global__ __launch_bounds__(256)
void flash_attn(const unsigned short* __restrict__ qg,
                const unsigned short* __restrict__ kg,
                const unsigned short* __restrict__ vtg,
                unsigned short* __restrict__ yg)
{
  __shared__ __align__(16) unsigned short lQ[64 * 128];  // 16 KB
  __shared__ __align__(16) float          lS[64 * 128];  // 32 KB
  __shared__ __align__(16) unsigned short lP[64 * 128];  // 16 KB
  __shared__ float lM[64], lL[64], lAl[64];

  const int tid  = threadIdx.x;
  const int lane = tid & 31;
  const int wave = tid >> 5;
  const int wm   = wave & 1;          // 2 M-blocks of 32 rows
  const int wn   = wave >> 1;         // 4 N-blocks of 32 cols
  const int qb = blockIdx.x, h = blockIdx.y, b = blockIdx.z;
  const int q0 = qb * 64;
  const int kvh = h >> 2;             // H / KVH = 4
  const int kh2 = lane >> 4;

  const unsigned short* qh  = qg  + ((size_t)(b * GH + h) * GT + q0) * GD;
  const unsigned short* khp = kg  + (size_t)(b * GKVH + kvh) * GT * GD;
  const unsigned short* vtp = vtg + (size_t)(b * GKVH + kvh) * GD * GT;

  // load Q tile: 64*128 bf16 = 1024 x 16B, 4 per thread
  for (int i = 0; i < 4; ++i) {
    int u  = tid * 4 + i;
    int r  = u >> 4;
    int cp = u & 15;
    *(u32x4*)&lQ[r * 128 + cp * 8] = *(const u32x4*)&qh[(size_t)r * GD + cp * 8];
  }
  if (tid < 64) { lM[tid] = -__builtin_inff(); lL[tid] = 0.f; }
  __syncthreads();

  v8f oacc[2][2];
  for (int a = 0; a < 2; ++a) for (int c = 0; c < 2; ++c) oacc[a][c] = v8zero();

  const int jend = (q0 + 63) / 128;
  for (int j = 0; j <= jend; ++j) {
    // ---- S = Q K^T (Q already scaled by 1/sqrt(D)) ----
    v8f sacc[2][2];
    for (int a = 0; a < 2; ++a) for (int c = 0; c < 2; ++c) sacc[a][c] = v8zero();

    for (int kk = 0; kk < 4; ++kk) {            // K loop over D = 128
      Frag16 fa[2], fb[2];
      for (int mt = 0; mt < 2; ++mt) {
        int r = wm * 32 + mt * 16 + (lane & 15);
        fa[mt].u4[0] = *(const u32x4*)&lQ[r * 128 + kk * 32 + kh2 * 16];
        fa[mt].u4[1] = *(const u32x4*)&lQ[r * 128 + kk * 32 + kh2 * 16 + 8];
      }
      for (int nt = 0; nt < 2; ++nt) {
        int c = j * 128 + wn * 32 + nt * 16 + (lane & 15);
        const unsigned short* bp = khp + (size_t)c * GD + kk * 32 + kh2 * 16;
        fb[nt].u4[0] = *(const u32x4*)bp;
        fb[nt].u4[1] = *(const u32x4*)(bp + 8);
      }
      for (int mt = 0; mt < 2; ++mt)
        for (int nt = 0; nt < 2; ++nt)
          sacc[mt][nt] = __builtin_amdgcn_wmma_f32_16x16x32_bf16(
              false, fa[mt].bf, false, fb[nt].bf, (short)0, sacc[mt][nt],
              false, false);
    }

    // ---- causal mask + spill S to LDS ----
    for (int mt = 0; mt < 2; ++mt)
      for (int nt = 0; nt < 2; ++nt)
        for (int i = 0; i < 8; ++i) {
          int row = wm * 32 + mt * 16 + (lane >> 4) * 8 + i;
          int col = wn * 32 + nt * 16 + (lane & 15);
          float v = sacc[mt][nt][i];
          if (j * 128 + col > q0 + row) v = -__builtin_inff();
          lS[row * 128 + col] = v;
        }
    __syncthreads();

    // ---- online softmax: 4 lanes per row, 32 cols each ----
    {
      int r = tid >> 2, part = tid & 3;
      const int cb = part * 32;
      float mold = lM[r];
      float mx = -__builtin_inff();
      for (int c = 0; c < 32; ++c) mx = fmaxf(mx, lS[r * 128 + cb + c]);
      mx = fmaxf(mx, __shfl_xor(mx, 1));
      mx = fmaxf(mx, __shfl_xor(mx, 2));
      mx = fmaxf(mx, mold);
      float sum = 0.f;
      for (int c = 0; c < 32; ++c) {
        float p = __expf(lS[r * 128 + cb + c] - mx);
        lP[r * 128 + cb + c] = f2bf(p);
        sum += p;
      }
      sum += __shfl_xor(sum, 1);
      sum += __shfl_xor(sum, 2);
      if (part == 0) {
        float alpha = __expf(mold - mx);
        lL[r]  = lL[r] * alpha + sum;
        lM[r]  = mx;
        lAl[r] = alpha;
      }
    }
    __syncthreads();

    // ---- rescale O by alpha(row) ----
    for (int mt = 0; mt < 2; ++mt) {
      int rowb = wm * 32 + mt * 16 + (lane >> 4) * 8;
      for (int i = 0; i < 8; ++i) {
        float a = lAl[rowb + i];
        for (int nt = 0; nt < 2; ++nt) oacc[mt][nt][i] *= a;
      }
    }

    // ---- O += P @ V  (V stored transposed [D,T]) ----
    for (int kk = 0; kk < 4; ++kk) {            // K loop over 128 kv positions
      Frag16 fa[2], fb[2];
      for (int mt = 0; mt < 2; ++mt) {
        int r = wm * 32 + mt * 16 + (lane & 15);
        fa[mt].u4[0] = *(const u32x4*)&lP[r * 128 + kk * 32 + kh2 * 16];
        fa[mt].u4[1] = *(const u32x4*)&lP[r * 128 + kk * 32 + kh2 * 16 + 8];
      }
      for (int nt = 0; nt < 2; ++nt) {
        int d = wn * 32 + nt * 16 + (lane & 15);
        const unsigned short* bp = vtp + (size_t)d * GT + j * 128 + kk * 32 + kh2 * 16;
        fb[nt].u4[0] = *(const u32x4*)bp;
        fb[nt].u4[1] = *(const u32x4*)(bp + 8);
      }
      for (int mt = 0; mt < 2; ++mt)
        for (int nt = 0; nt < 2; ++nt)
          oacc[mt][nt] = __builtin_amdgcn_wmma_f32_16x16x32_bf16(
              false, fa[mt].bf, false, fb[nt].bf, (short)0, oacc[mt][nt],
              false, false);
    }
    __syncthreads();
  }

  // ---- normalize and write y [B,T,H,D] (== [B,T,C] for out-proj) ----
  for (int mt = 0; mt < 2; ++mt)
    for (int nt = 0; nt < 2; ++nt)
      for (int i = 0; i < 8; ++i) {
        int row = wm * 32 + mt * 16 + (lane >> 4) * 8 + i;
        int col = wn * 32 + nt * 16 + (lane & 15);
        float v = oacc[mt][nt][i] / lL[row];
        yg[((size_t)(b * GT + q0 + row) * GH + h) * GD + col] = f2bf(v);
      }
}

// =====================================================================
// launch
// =====================================================================
extern "C" void kernel_launch(void* const* d_in, const int* in_sizes, int n_in,
                              void* d_out, int out_size, void* d_ws, size_t ws_size,
                              hipStream_t stream) {
  const float* x    = (const float*)d_in[0];
  const float* Wqkv = (const float*)d_in[1];
  const float* bqkv = (const float*)d_in[2];
  const float* Wo   = (const float*)d_in[3];
  const float* bo   = (const float*)d_in[4];
  float* out = (float*)d_out;

  // workspace carve-out (~252 MB)
  char* ws = (char*)d_ws;
  size_t off = 0;
  auto carve = [&](size_t bytes) {
    char* p = ws + off;
    off += (bytes + 255) & ~(size_t)255;
    return p;
  };
  unsigned short* xb    = (unsigned short*)carve((size_t)GB * GT * GC * 2);
  unsigned short* wqkvb = (unsigned short*)carve((size_t)QKVO * GC * 2);
  unsigned short* wob   = (unsigned short*)carve((size_t)GC * GC * 2);
  unsigned short* qkvb  = (unsigned short*)carve((size_t)GB * GT * QKVO * 2);
  unsigned short* qb    = (unsigned short*)carve((size_t)GB * GH * GT * GD * 2);
  unsigned short* kb    = (unsigned short*)carve((size_t)GB * GKVH * GT * GD * 2);
  unsigned short* vtb   = (unsigned short*)carve((size_t)GB * GKVH * GD * GT * 2);
  unsigned short* yb    = (unsigned short*)carve((size_t)GB * GT * GC * 2);

  // 1) fp32 -> bf16 conversions
  cvt_f32_bf16<<<(GB * GT * GC) / 1024, 256, 0, stream>>>(x, xb);
  cvt_f32_bf16<<<(QKVO * GC) / 1024, 256, 0, stream>>>(Wqkv, wqkvb);
  cvt_f32_bf16<<<(GC * GC) / 1024, 256, 0, stream>>>(Wo, wob);

  // 2) QKV projection: [4096,4096] @ [6144,4096]^T -> bf16 [B*T, 6144]
  {
    dim3 g(QKVO / 128, (GB * GT) / 128);
    gemm_bf16<<<g, 256, 0, stream>>>(xb, wqkvb, bqkv, qkvb,
                                     GB * GT, QKVO, GC, /*out_bf16=*/1);
  }

  // 3) RoPE + scatter to q/k/vT
  rope_scatter<<<(GB * GT * 3072) / 256, 256, 0, stream>>>(qkvb, qb, kb, vtb);

  // 4) causal GQA flash attention -> y bf16 [B,T,C]
  flash_attn<<<dim3(GT / 64, GH, GB), 256, 0, stream>>>(qb, kb, vtb, yb);

  // 5) output projection: y @ Wo^T + bo -> fp32 d_out
  {
    dim3 g(GC / 128, (GB * GT) / 128);
    gemm_bf16<<<g, 256, 0, stream>>>(yb, wob, bo, out,
                                     GB * GT, GC, GC, /*out_bf16=*/0);
  }
}
